// MultiHeadAttention_48687749267513
// MI455X (gfx1250) — matrix operand order repaired and checked by reading, not compiled, not run on hardware
//
#include <hip/hip_runtime.h>

// ---------------------------------------------------------------------------
// MI455X / gfx1250 fused multi-head attention, bf16 WMMA path.
// All matmuls -> v_wmma_f32_16x16x32_bf16 (fp32 accumulate).
// B-operands stored N-major (K contiguous): every fragment is b128 loads.
// GEMM waves compute 32x64 (2 A-frags x 4 shared B-frags = 8 WMMA/step).
// Flash softmax branchless; mask bytes loaded in-loop to keep VGPR live
// ranges short (no spills of the Q fragments).
// ---------------------------------------------------------------------------

#define DEVINL __device__ __forceinline__

typedef __attribute__((ext_vector_type(16))) __bf16 bf16x16;
typedef __attribute__((ext_vector_type(8)))  __bf16 bf16x8;
typedef __attribute__((ext_vector_type(8)))  float  floatx8;

constexpr int Bc  = 4;
constexpr int Lc  = 1024;
constexpr int Dc  = 1024;
constexpr int Hc  = 16;
constexpr int DKc = 64;
constexpr int DVc = 64;
constexpr int HDV = Hc * DVc;   // 1024

// ---- helpers --------------------------------------------------------------

DEVINL __bf16 f32_to_bf16(float f) {                 // round-to-nearest-even
  unsigned u   = __builtin_bit_cast(unsigned, f);
  unsigned lsb = (u >> 16) & 1u;
  u += 0x7fffu + lsb;
  unsigned short s = (unsigned short)(u >> 16);
  return __builtin_bit_cast(__bf16, s);
}

DEVINL bf16x16 cat8(bf16x8 lo, bf16x8 hi) {
  return __builtin_shufflevector(lo, hi, 0,1,2,3,4,5,6,7,8,9,10,11,12,13,14,15);
}

DEVINL floatx8 wmma_bf16f32(bf16x16 a, bf16x16 b, floatx8 c) {
  // D = A(16x32 bf16) * B(32x16 bf16) + C(16x16 f32)
  return __builtin_amdgcn_wmma_f32_16x16x32_bf16(
      /*neg_a=*/false, a, /*neg_b=*/false, b,
      /*c_mod=*/(short)0, c, /*reuse_a=*/false, /*reuse_b=*/false);
}

// A-fragment from row-major [*, lda] bf16: rows m0..m0+15, K = k0..k0+31.
// Lane = row (mod 16); lane-half selects K sub-bands {0..7,16..23}/{8..15,24..31}.
DEVINL bf16x16 load_a_frag(const __bf16* A, int lda, int m0, int k0, int lane) {
  const int hsel = lane >> 4;
  const int r    = m0 + (lane & 15);
  const __bf16* p = A + (size_t)r * lda + k0 + hsel * 8;
  return cat8(*(const bf16x8*)(p), *(const bf16x8*)(p + 16));
}

// B-fragment from N-major (transposed) storage Bt[N][K]: cols n0..n0+15,
// K = k0..k0+31.  Lane = col (mod 16); lane-half selects K sub-bands {0..7,16..23}/{8..15,24..31}.
// Per lane: two 8-element runs -> two b128 loads.
DEVINL bf16x16 load_bt_frag(const __bf16* Bt, int ldk, int n0, int k0, int lane) {
  const __bf16* p = Bt + (size_t)(n0 + (lane & 15)) * ldk + k0 + (lane >> 4) * 8;
  return cat8(*(const bf16x8*)(p), *(const bf16x8*)(p + 16));
}

// ---- fp32 -> bf16 conversion ----------------------------------------------

__global__ void cvt_f32_to_bf16(const float* __restrict__ src,
                                __bf16* __restrict__ dst, int n) {
  int i = blockIdx.x * blockDim.x + threadIdx.x;
  if (i < n) dst[i] = f32_to_bf16(src[i]);
}

// batched transpose+convert: src [z][R][C] f32 -> dst [z][C][R] bf16.
// Indexed over the output so writes are coalesced; reads hit L2.
__global__ void cvt_transpose_f32_to_bf16(const float* __restrict__ src,
                                          __bf16* __restrict__ dst,
                                          int R, int C) {
  const size_t base = (size_t)blockIdx.z * R * C;
  int o = blockIdx.x * blockDim.x + threadIdx.x;      // o = c*R + r
  if (o < R * C) {
    int c = o / R, r = o % R;
    dst[base + o] = f32_to_bf16(src[base + (size_t)r * C + c]);
  }
}

// ---- bf16 GEMM: C[M,N] = A[M,K] * Bt[N,K]^T -------------------------------
// 256 threads = 8 waves; wave => 32 rows x 64 cols (2x4 WMMA tiles,
// B-fragments shared across the two row tiles). Block covers 256 rows.
// Pure b128 traffic, no LDS/barriers; reuse via 192MB L2.
// grid = (M/256, N/64, batch); z: A += (z/Hdim)*sA, Bt += (z%Hdim)*sB,
// C += z*sC.  OUT_TRANS stores C as [N,M] (ldc = M-stride) with b128 stores.

template <bool OUT_BF16, bool OUT_TRANS>
__global__ __launch_bounds__(256) void gemm_bf16(
    const __bf16* __restrict__ A, int lda, size_t sA,
    const __bf16* __restrict__ Bt, size_t sB,
    void* __restrict__ Cp, int ldc, size_t sC,
    int K, int Hdim) {
  const int z = blockIdx.z;
  A  += (size_t)(z / Hdim) * sA;
  Bt += (size_t)(z % Hdim) * sB;

  const int n0   = blockIdx.y * 64;
  const int wave = threadIdx.x >> 5;
  const int lane = threadIdx.x & 31;
  const int m0   = blockIdx.x * 256 + wave * 32;
  const int hsel = lane >> 4;
  const int colL = lane & 15;

  floatx8 acc[2][4];
#pragma unroll
  for (int r = 0; r < 2; ++r)
#pragma unroll
    for (int t = 0; t < 4; ++t) acc[r][t] = {};

  const __bf16* arow0 = A + (size_t)(m0 + colL) * lda + hsel * 8;
  const __bf16* arow1 = arow0 + (size_t)16 * lda;
  const __bf16* brow[4];
#pragma unroll
  for (int t = 0; t < 4; ++t)
    brow[t] = Bt + (size_t)(n0 + t * 16 + colL) * K + hsel * 8;

#pragma unroll 2
  for (int k0 = 0; k0 < K; k0 += 32) {
    __builtin_prefetch(arow0 + k0 + 64, 0, 1);   // global_prefetch_b8
    bf16x16 a0 = cat8(*(const bf16x8*)(arow0 + k0),
                      *(const bf16x8*)(arow0 + k0 + 16));
    bf16x16 a1 = cat8(*(const bf16x8*)(arow1 + k0),
                      *(const bf16x8*)(arow1 + k0 + 16));
#pragma unroll
    for (int t = 0; t < 4; ++t) {
      bf16x16 bb = cat8(*(const bf16x8*)(brow[t] + k0),
                        *(const bf16x8*)(brow[t] + k0 + 16));
      acc[0][t] = wmma_bf16f32(a0, bb, acc[0][t]);
      acc[1][t] = wmma_bf16f32(a1, bb, acc[1][t]);
    }
  }

#pragma unroll
  for (int r = 0; r < 2; ++r) {
    const int mb = m0 + r * 16 + hsel * 8;       // C tile: vgpr=row, lane=col
    if constexpr (OUT_TRANS) {                   // [N,M]: lane owns 8 contig M
      __bf16* C = (__bf16*)Cp + (size_t)z * sC;
#pragma unroll
      for (int t = 0; t < 4; ++t) {
        bf16x8 v;
#pragma unroll
        for (int j = 0; j < 8; ++j) v[j] = f32_to_bf16(acc[r][t][j]);
        *(bf16x8*)&C[(size_t)(n0 + t * 16 + colL) * ldc + mb] = v;
      }
    } else if constexpr (OUT_BF16) {
      __bf16* C = (__bf16*)Cp + (size_t)z * sC;
#pragma unroll
      for (int t = 0; t < 4; ++t)
#pragma unroll
        for (int j = 0; j < 8; ++j)
          C[(size_t)(mb + j) * ldc + n0 + t * 16 + colL] = f32_to_bf16(acc[r][t][j]);
    } else {
      float* C = (float*)Cp + (size_t)z * sC;
#pragma unroll
      for (int t = 0; t < 4; ++t)
#pragma unroll
        for (int j = 0; j < 8; ++j)
          C[(size_t)(mb + j) * ldc + n0 + t * 16 + colL] = acc[r][t][j];
    }
  }
}

// ---- flash attention ------------------------------------------------------
// q,k row-major [B*H, L, 64]; v transposed [B*H, 64, L].
// block = 128 thr = 4 waves; wave owns 16 query rows; 32-key blocks.
// S tiles via wmma(qA, k^T-B); online softmax in C layout (16-lane xor
// reductions, branchless masking); P re-laid C->A through per-wave LDS;
// ctx += wmma(P, Vt).

__global__ __launch_bounds__(128, 1) void flash_attn(
    const __bf16* __restrict__ qg,
    const __bf16* __restrict__ kg,
    const __bf16* __restrict__ vtg,
    const unsigned char* __restrict__ minfo,
    __bf16* __restrict__ ctx) {
  __shared__ __align__(16) __bf16 Pls[4][16][40];  // 16x32 P tile / wave, padded

  const int wave = threadIdx.x >> 5;
  const int lane = threadIdx.x & 31;
  const int bh   = blockIdx.y;
  const int b    = bh / Hc;
  const int h    = bh % Hc;
  const int m0   = blockIdx.x * 64 + wave * 16;
  const int hsel = lane >> 4;
  const int colL = lane & 15;
  const int rowb = m0 + hsel * 8;                 // rows this lane's 8 regs hold

  const __bf16* qp = qg  + (size_t)bh * Lc * DKc;
  const __bf16* kp = kg  + (size_t)bh * Lc * DKc;
  const __bf16* vp = vtg + (size_t)bh * DVc * Lc; // [64][L]
  const unsigned char* mp = minfo + (size_t)b * Lc * Lc;

  const bf16x16 aq0 = load_a_frag(qp, DKc, m0, 0,  lane);
  const bf16x16 aq1 = load_a_frag(qp, DKc, m0, 32, lane);

  floatx8 acc[4];
#pragma unroll
  for (int t = 0; t < 4; ++t) acc[t] = {};
  float mrow[8], lrow[8];
#pragma unroll
  for (int j = 0; j < 8; ++j) { mrow[j] = -3.0e38f; lrow[j] = 0.f; }

  for (int ks = 0; ks < m0 + 16; ks += 32) {      // causal: keys <= m0+15
    // ---- S = q . k^T : k row-major acts as N-major B (keys = N) ----
    floatx8 s0 = {}, s1 = {};
    s0 = wmma_bf16f32(aq0, load_bt_frag(kp, DKc, ks,      0,  lane), s0);
    s0 = wmma_bf16f32(aq1, load_bt_frag(kp, DKc, ks,      32, lane), s0);
    s1 = wmma_bf16f32(aq0, load_bt_frag(kp, DKc, ks + 16, 0,  lane), s1);
    s1 = wmma_bf16f32(aq1, load_bt_frag(kp, DKc, ks + 16, 32, lane), s1);

    // ---- online softmax (rows live across one 16-lane half) ----
    const int key0 = ks + colL;
    const int key1 = ks + 16 + colL;
#pragma unroll
    for (int j = 0; j < 8; ++j) {
      const int row = rowb + j;
      const size_t roff = (size_t)row * Lc;
      // unconditional mask loads (always in-bounds), bitwise combine ->
      // branchless; short live range keeps aq0/aq1 resident (no spills)
      const bool k0m = (key0 > row) | (mp[roff + key0] != 0);
      const bool k1m = (key1 > row) | (mp[roff + key1] != 0);
      const float v0 = k0m ? -3.0e38f : s0[j] * 0.125f; // 1/sqrt(64)
      const float v1 = k1m ? -3.0e38f : s1[j] * 0.125f;
      float x = fmaxf(v0, v1);
      x = fmaxf(x, __shfl_xor(x, 1));
      x = fmaxf(x, __shfl_xor(x, 2));
      x = fmaxf(x, __shfl_xor(x, 4));
      x = fmaxf(x, __shfl_xor(x, 8));
      const float mnew = fmaxf(mrow[j], x);
      const float p0 = k0m ? 0.f : __expf(v0 - mnew);
      const float p1 = k1m ? 0.f : __expf(v1 - mnew);
      float srow = p0 + p1;
      srow += __shfl_xor(srow, 1);
      srow += __shfl_xor(srow, 2);
      srow += __shfl_xor(srow, 4);
      srow += __shfl_xor(srow, 8);
      const float al = __expf(mrow[j] - mnew);
      lrow[j] = lrow[j] * al + srow;
      mrow[j] = mnew;
      acc[0][j] *= al;                            // inline rescale: no alv[]
      acc[1][j] *= al;
      acc[2][j] *= al;
      acc[3][j] *= al;
      Pls[wave][hsel * 8 + j][colL]      = f32_to_bf16(p0);
      Pls[wave][hsel * 8 + j][16 + colL] = f32_to_bf16(p1);
    }

    // ---- P as A-frag (C->A relayout via per-wave LDS) ----
    const __bf16* pp = &Pls[wave][colL][hsel * 8];
    bf16x16 pa = cat8(*(const bf16x8*)pp, *(const bf16x8*)(pp + 16));

    // ---- ctx += P . V : Vt[dv][key] is N-major B (dv = N, keys = K) ----
#pragma unroll
    for (int t = 0; t < 4; ++t)
      acc[t] = wmma_bf16f32(pa, load_bt_frag(vp, Lc, t * 16, ks, lane), acc[t]);
  }

  // ---- normalize + store head-major ctx [B, L, H*DV] bf16 ----
#pragma unroll
  for (int j = 0; j < 8; ++j) {
    const float inv = lrow[j] > 0.f ? 1.f / lrow[j] : 0.f;  // fully masked -> 0
    const int row = rowb + j;
    __bf16* cp = ctx + ((size_t)b * Lc + row) * HDV + h * DVc;
#pragma unroll
    for (int t = 0; t < 4; ++t)
      cp[t * 16 + colL] = f32_to_bf16(acc[t][j] * inv);
  }
}

// ---------------------------------------------------------------------------

extern "C" void kernel_launch(void* const* d_in, const int* in_sizes, int n_in,
                              void* d_out, int out_size, void* d_ws, size_t ws_size,
                              hipStream_t stream) {
  const float* Q  = (const float*)d_in[0];
  const float* K  = (const float*)d_in[1];
  const float* V  = (const float*)d_in[2];
  const unsigned char* MI = (const unsigned char*)d_in[3];
  const float* Wq = (const float*)d_in[4];
  const float* Wk = (const float*)d_in[5];
  const float* Wv = (const float*)d_in[6];
  const float* Wo = (const float*)d_in[7];

  constexpr size_t nQKV = (size_t)Bc * Lc * Dc;        // 4 Mi elems
  constexpr size_t nW   = (size_t)Hc * Dc * DKc;       // 1 Mi
  constexpr size_t nWo  = (size_t)HDV * Dc;            // 1 Mi
  constexpr size_t nH   = (size_t)Bc * Hc * Lc * DKc;  // 4 Mi
  constexpr size_t nCtx = (size_t)Bc * Lc * HDV;       // 4 Mi

  char* ws = (char*)d_ws;
  size_t off = 0;
  auto carve = [&](size_t elems) {
    __bf16* p = (__bf16*)(ws + off);
    off = (off + elems * 2 + 255) & ~(size_t)255;
    return p;
  };
  __bf16* Qb   = carve(nQKV);
  __bf16* Kb   = carve(nQKV);
  __bf16* Vb   = carve(nQKV);
  __bf16* Wqt  = carve(nW);    // [H][DK][D]
  __bf16* Wkt  = carve(nW);
  __bf16* Wvt  = carve(nW);
  __bf16* Wot  = carve(nWo);   // [D][HDV]
  __bf16* qh   = carve(nH);    // [B*H][L][DK]
  __bf16* kh   = carve(nH);    // [B*H][L][DK]
  __bf16* vt   = carve(nH);    // [B*H][DV][L]   (transposed)
  __bf16* ctxb = carve(nCtx);  // [B][L][H*DV]

  auto cvt = [&](const float* s, __bf16* d, size_t n) {
    cvt_f32_to_bf16<<<dim3((unsigned)((n + 255) / 256)), dim3(256), 0, stream>>>(
        s, d, (int)n);
  };
  cvt(Q, Qb, nQKV); cvt(K, Kb, nQKV); cvt(V, Vb, nQKV);

  // weights: convert + transpose (B operands want N-major / K-contiguous)
  {
    dim3 g((Dc * DKc + 255) / 256, 1, Hc);
    cvt_transpose_f32_to_bf16<<<g, 256, 0, stream>>>(Wq, Wqt, Dc, DKc);
    cvt_transpose_f32_to_bf16<<<g, 256, 0, stream>>>(Wk, Wkt, Dc, DKc);
    cvt_transpose_f32_to_bf16<<<g, 256, 0, stream>>>(Wv, Wvt, Dc, DVc);
    dim3 go((HDV * Dc + 255) / 256, 1, 1);
    cvt_transpose_f32_to_bf16<<<go, 256, 0, stream>>>(Wo, Wot, HDV, Dc);
  }

  // per-(b,h) projections: [L x D] * [D x 64]
  dim3 gp(Lc / 256, DKc / 64, Bc * Hc);
  gemm_bf16<true, false><<<gp, 256, 0, stream>>>(
      Qb, Dc, (size_t)Lc * Dc, Wqt, (size_t)DKc * Dc,
      qh, DKc, (size_t)Lc * DKc, Dc, Hc);
  gemm_bf16<true, false><<<gp, 256, 0, stream>>>(
      Kb, Dc, (size_t)Lc * Dc, Wkt, (size_t)DKc * Dc,
      kh, DKc, (size_t)Lc * DKc, Dc, Hc);
  gemm_bf16<true, true><<<gp, 256, 0, stream>>>(     // V stored transposed
      Vb, Dc, (size_t)Lc * Dc, Wvt, (size_t)DVc * Dc,
      vt, Lc, (size_t)DVc * Lc, Dc, Hc);

  // flash attention -> ctx [B, L, H*DV] bf16
  dim3 ga(Lc / 64, Bc * Hc);
  flash_attn<<<ga, 128, 0, stream>>>(qh, kh, vt, MI, ctxb);

  // output projection: [B*L x 1024] * [1024 x 1024] -> f32 d_out
  dim3 go2((Bc * Lc) / 256, Dc / 64, 1);
  gemm_bf16<false, false><<<go2, 256, 0, stream>>>(
      ctxb, HDV, 0, Wot, 0, d_out, Dc, 0, HDV, 1);
}
